// STNet_Point_Classifier_Advanced_82910048682148
// MI455X (gfx1250) — matrix operand order repaired and checked by reading, not compiled, not run on hardware
//
#include <hip/hip_runtime.h>

typedef __attribute__((ext_vector_type(16))) __bf16 v16bf;
typedef __attribute__((ext_vector_type(8)))  __bf16 v8bf;
typedef __attribute__((ext_vector_type(4)))  __bf16 v4bf;
typedef __attribute__((ext_vector_type(8)))  float  v8f;

// ---------------------------------------------------------------------------
// Problem constants (from reference)
// ---------------------------------------------------------------------------
#define Bc  2
#define Np  2048
#define Cc  256
#define Hh  8
#define HDc 32
#define LDT 40   // LDS row stride (elements): 80 B -> 16B-aligned rows,
                 // per-lane b128 bank bases r*20 mod 64 all distinct.

// Build a v16bf fragment from two contiguous 8x bf16 runs (ds_load_b128 x2).
__device__ inline v16bf ld_frag2(const __bf16* p0, const __bf16* p1) {
  union { v16bf v; v8bf h[2]; } u;
  u.h[0] = *(const v8bf*)p0;
  u.h[1] = *(const v8bf*)p1;
  return u.v;
}

// ---------------------------------------------------------------------------
// LayerNorm over channel dim of [B,C,N]; one wave per (b,n) column.
// ---------------------------------------------------------------------------
__global__ __launch_bounds__(256) void ln_cn_kernel(
    const float* __restrict__ X, const float* __restrict__ g,
    const float* __restrict__ bvec, float* __restrict__ Y,
    long x_bs, long y_bs, int total_cols)
{
  const int wave = blockIdx.x * 8 + (threadIdx.x >> 5);
  const int lane = threadIdx.x & 31;
  if (wave >= total_cols) return;
  const int b = wave / Np, n = wave % Np;
  const float* Xb = X + (size_t)b * x_bs + n;

  float vals[8];
  float s = 0.f;
#pragma unroll
  for (int i = 0; i < 8; ++i) {
    vals[i] = Xb[(size_t)(lane + 32 * i) * Np];
    s += vals[i];
  }
#pragma unroll
  for (int m = 16; m >= 1; m >>= 1) s += __shfl_xor(s, m);
  const float mean = s * (1.0f / (float)Cc);
  float vs = 0.f;
#pragma unroll
  for (int i = 0; i < 8; ++i) { float d = vals[i] - mean; vs += d * d; }
#pragma unroll
  for (int m = 16; m >= 1; m >>= 1) vs += __shfl_xor(vs, m);
  const float rstd = rsqrtf(vs * (1.0f / (float)Cc) + 1e-5f);

  float* Yb = Y + (size_t)b * y_bs + n;
#pragma unroll
  for (int i = 0; i < 8; ++i) {
    const int c = lane + 32 * i;
    Yb[(size_t)c * Np] = (vals[i] - mean) * rstd * g[c] + bvec[c];
  }
}

// ---------------------------------------------------------------------------
// Squared L2 norm over channels of [B,C,N]: out[b*N+n] = sum_c X[b,c,n]^2
// ---------------------------------------------------------------------------
__global__ __launch_bounds__(256) void rownorm2_kernel(
    const float* __restrict__ X, float* __restrict__ out, int total_cols)
{
  const int wave = blockIdx.x * 8 + (threadIdx.x >> 5);
  const int lane = threadIdx.x & 31;
  if (wave >= total_cols) return;
  const int b = wave / Np, n = wave % Np;
  const float* Xb = X + (size_t)b * Cc * Np + n;
  float s = 0.f;
#pragma unroll
  for (int i = 0; i < 8; ++i) {
    float v = Xb[(size_t)(lane + 32 * i) * Np];
    s += v * v;
  }
#pragma unroll
  for (int m = 16; m >= 1; m >>= 1) s += __shfl_xor(s, m);
  if (lane == 0) out[wave] = s;
}

// ---------------------------------------------------------------------------
// bf16 WMMA GEMM, compile-time specialized.
//   TRANSA=0: A(m,k) = A[m*lda + k]   (weights, row-major)
//   TRANSA=1: A(m,k) = A[k*lda + m]   (qd^T for feature-space cdist)
//   B(k,n) = B[k*ldb + n]  (ldb-major activations; unit n-stride assumed)
//   Y[b][m,n] = act( dot * scale[m]*smul + shift[m] ) + R[b][m,n]
// Block = 256 thr = 8 waves; tile 64(M) x 32(N); K step 32; one 16x16 WMMA
// per wave per step. Requires M%64==0, Nn%32==0, K%32==0, lda%4==0, ldb%4==0.
// Staging: global float4 loads; fragments: conflict-free ds_load_b128.
// ---------------------------------------------------------------------------
template<int TRANSA, int HAS_SCALE, int HAS_SHIFT, int HAS_RES, int RELU>
__global__ __launch_bounds__(256) void gemm_bf16_wmma(
    const float* __restrict__ A, const float* __restrict__ Bm,
    float* __restrict__ Y,
    const float* __restrict__ scale, const float* __restrict__ shift,
    const float* __restrict__ R,
    int M, int Nn, int K,
    long lda, long a_bs, long ldb, long b_bs, long y_bs, long r_bs,
    float smul)
{
  __shared__ __bf16 As[64][LDT];   // [m][k]
  __shared__ __bf16 Bs[32][LDT];   // [n][k]  (LDS-transposed for b128 frags)

  const int tid  = threadIdx.x;
  const int bz   = blockIdx.z;
  const int n0   = blockIdx.x * 32;
  const int m0   = blockIdx.y * 64;
  const int lane = tid & 31;
  const int wave = tid >> 5;
  const int half = lane >> 4;
  const int l16  = lane & 15;
  const int wm   = wave & 3;
  const int wn   = wave >> 2;

  const float* Ab = A  + (size_t)bz * a_bs;
  const float* Bb = Bm + (size_t)bz * b_bs;

  v8f acc = {};

  for (int k0 = 0; k0 < K; k0 += 32) {
    if constexpr (!TRANSA) {
      if (k0 + 32 < K)
        __builtin_prefetch(&Ab[(size_t)m0 * lda + (k0 + 32)], 0, 1);
      // A 64x32: 512 float4 loads, 2/thread, store v4bf (ds_store_b64)
#pragma unroll
      for (int t = 0; t < 2; ++t) {
        const int l = tid + t * 256;
        const int mi = l >> 3, ki4 = (l & 7) * 4;
        const float4 f = *(const float4*)(Ab + (size_t)(m0 + mi) * lda + (k0 + ki4));
        v4bf b4;
        b4[0] = (__bf16)f.x; b4[1] = (__bf16)f.y;
        b4[2] = (__bf16)f.z; b4[3] = (__bf16)f.w;
        *(v4bf*)&As[mi][ki4] = b4;
      }
    } else {
      // A(m,k)=A[k*lda+m]: coalesce along m, scatter into [m][k]
#pragma unroll
      for (int t = 0; t < 2; ++t) {
        const int l = tid + t * 256;
        const int ki = l >> 4, mi4 = (l & 15) * 4;
        const float4 f = *(const float4*)(Ab + (size_t)(k0 + ki) * lda + (m0 + mi4));
        As[mi4 + 0][ki] = (__bf16)f.x;
        As[mi4 + 1][ki] = (__bf16)f.y;
        As[mi4 + 2][ki] = (__bf16)f.z;
        As[mi4 + 3][ki] = (__bf16)f.w;
      }
    }
    {
      // B 32x32: 256 float4 loads, 1/thread, transpose into Bs[n][k]
      const int ki = tid >> 3, ni4 = (tid & 7) * 4;
      const float4 f = *(const float4*)(Bb + (size_t)(k0 + ki) * ldb + (n0 + ni4));
      Bs[ni4 + 0][ki] = (__bf16)f.x;
      Bs[ni4 + 1][ki] = (__bf16)f.y;
      Bs[ni4 + 2][ki] = (__bf16)f.z;
      Bs[ni4 + 3][ki] = (__bf16)f.w;
    }
    __syncthreads();

    const __bf16* arow = &As[wm * 16 + l16][0];
    const __bf16* brow = &Bs[wn * 16 + l16][0];
    // A frag: lane half owns k = {half*8..+7, 16+half*8..+7}
    const v16bf af  = ld_frag2(arow + half * 8, arow + 16 + half * 8);
    // B frag: lane half owns k = {half*8..+7, 16+half*8..+7}
    const v16bf bfr = ld_frag2(brow + half * 8, brow + 16 + half * 8);
    acc = __builtin_amdgcn_wmma_f32_16x16x32_bf16(false, af, false, bfr,
                                                  (short)0, acc, false, false);
    __syncthreads();
  }

  // epilogue: acc element j at (M = j + 8*half, N = l16) of the wave tile
#pragma unroll
  for (int j = 0; j < 8; ++j) {
    const int row = m0 + wm * 16 + half * 8 + j;
    const int col = n0 + wn * 16 + l16;
    float v = acc[j];
    if constexpr (HAS_SCALE) v *= scale[row] * smul;
    if constexpr (HAS_SHIFT) v += shift[row];
    if constexpr (RELU)      v = fmaxf(v, 0.0f);
    if constexpr (HAS_RES)   v += R[(size_t)bz * r_bs + (size_t)row * Nn + col];
    Y[(size_t)bz * y_bs + (size_t)row * Nn + col] = v;
  }
}

// ---------------------------------------------------------------------------
// Fused dual-path flash attention (grid: N/16 x H x B; 1 wave per block).
//   sim    = q.k^T/sqrt(HD) + spatial_bias(xyz)          (WMMA per 16 keys)
//   dissim = sqrt(|qd|^2+|kd|^2-2 qd.kd) + spatial_bias  (precomputed dot)
// Online softmax on both paths; P -> LDS -> b128 A-fragments; ctx += P.V.
// ---------------------------------------------------------------------------
__global__ __launch_bounds__(32) void attn_dual_kernel(
    const float* __restrict__ qb, const float* __restrict__ kbuf,
    const float* __restrict__ vbuf,
    const float* __restrict__ dotd, const float* __restrict__ nqv,
    const float* __restrict__ nkv,
    const float* __restrict__ xyzA, const float* __restrict__ xyzB,
    const float* __restrict__ sb_w1, const float* __restrict__ sb_bn_g,
    const float* __restrict__ sb_bn_b, const float* __restrict__ sb_w2,
    const float* __restrict__ sb_b2,
    float* __restrict__ ctx_s, float* __restrict__ ctx_d)
{
  __shared__ __bf16 Ps[16][LDT];
  __shared__ __bf16 Pd[16][LDT];

  const int b    = blockIdx.z;
  const int h    = blockIdx.y;
  const int n0   = blockIdx.x * 16;
  const int lane = threadIdx.x;
  const int half = lane >> 4;
  const int l16  = lane & 15;

  const size_t CN = (size_t)Cc * Np;
  const float* qp  = qb   + (size_t)b * CN + (size_t)h * HDc * Np;
  const float* kp  = kbuf + (size_t)b * CN + (size_t)h * HDc * Np;
  const float* vp  = vbuf + (size_t)b * CN + (size_t)h * HDc * Np;
  const float* dd  = dotd + (size_t)b * Np * Np;
  const float* nqp = nqv + (size_t)b * Np;
  const float* nkp = nkv + (size_t)b * Np;
  const float* xa  = xyzA + (size_t)b * Np * 3;
  const float* xbp = xyzB + (size_t)b * Np * 3;

  // spatial-bias MLP coefficients for this head (BN folded)
  const float bnrs = 0.99999500003749978f;  // 1/sqrt(1+1e-5)
  float c1[4], c0[4], w2h[4];
#pragma unroll
  for (int jj = 0; jj < 4; ++jj) {
    c1[jj]  = -sb_w1[jj] * sb_bn_g[jj] * bnrs;
    c0[jj]  =  sb_bn_b[jj];
    w2h[jj] =  sb_w2[h * 4 + jj];
  }
  const float bias0 = sb_b2[h];

  // q A-fragment with 1/sqrt(HD) folded in
  const float invs = 0.176776695296637f;
  v16bf aq;
  {
    const int n = n0 + l16;
#pragma unroll
    for (int i = 0; i < 16; ++i) {
      const int kd = (i < 8 ? i : i + 8) + half * 8;
      aq[i] = (__bf16)(qp[(size_t)kd * Np + n] * invs);
    }
  }

  // per-lane row constants (rows j + 8*half, matching C-fragment layout)
  float ax[8], ay[8], az[8], nqr[8];
#pragma unroll
  for (int j = 0; j < 8; ++j) {
    const int n = n0 + half * 8 + j;
    ax[j] = xa[n * 3 + 0]; ay[j] = xa[n * 3 + 1]; az[j] = xa[n * 3 + 2];
    nqr[j] = nqp[n];
  }

  float rmax_s[8], rsum_s[8], rmax_d[8], rsum_d[8];
#pragma unroll
  for (int j = 0; j < 8; ++j) {
    rmax_s[j] = -1e30f; rsum_s[j] = 0.f;
    rmax_d[j] = -1e30f; rsum_d[j] = 0.f;
  }
  v8f acc_s0 = {}, acc_s1 = {}, acc_d0 = {}, acc_d1 = {};
  const v8f cz = {};

  for (int mc = 0; mc < Np; mc += 32) {
    float sS[2][8], sD[2][8];
#pragma unroll
    for (int t = 0; t < 2; ++t) {
      const int m = mc + t * 16 + l16;
      v16bf bk;
#pragma unroll
      for (int i = 0; i < 16; ++i) {           // coalesced across lanes per i
        const int kk = (i < 8 ? i : i + 8) + half * 8;
        bk[i] = (__bf16)(kp[(size_t)kk * Np + m]);
      }
      v8f S = __builtin_amdgcn_wmma_f32_16x16x32_bf16(false, aq, false, bk,
                                                      (short)0, cz, false, false);
      const float bx = xbp[m * 3 + 0], by = xbp[m * 3 + 1], bz3 = xbp[m * 3 + 2];
      const float nkm = nkp[m];
#pragma unroll
      for (int j = 0; j < 8; ++j) {
        const float dx = ax[j] - bx, dy = ay[j] - by, dz = az[j] - bz3;
        const float dist = sqrtf(dx * dx + dy * dy + dz * dz);
        float bias = bias0;
#pragma unroll
        for (int jj = 0; jj < 4; ++jj)
          bias += w2h[jj] * fmaxf(dist * c1[jj] + c0[jj], 0.f);
        sS[t][j] = S[j] + bias;
        const int nr = n0 + half * 8 + j;
        const float dot = dd[(size_t)nr * Np + m];
        const float d2 = nqr[j] + nkm - 2.f * dot;
        sD[t][j] = sqrtf(fmaxf(d2, 0.f)) + bias;
      }
    }

    // online softmax, both paths; row stats live in 16-lane shuffle groups
#pragma unroll
    for (int j = 0; j < 8; ++j) {
      const int row = half * 8 + j;
      {
        float cm = fmaxf(sS[0][j], sS[1][j]);
        cm = fmaxf(cm, __shfl_xor(cm, 1)); cm = fmaxf(cm, __shfl_xor(cm, 2));
        cm = fmaxf(cm, __shfl_xor(cm, 4)); cm = fmaxf(cm, __shfl_xor(cm, 8));
        const float nm = fmaxf(rmax_s[j], cm);
        const float corr = __expf(rmax_s[j] - nm);
        rmax_s[j] = nm;
        const float p0 = __expf(sS[0][j] - nm);
        const float p1 = __expf(sS[1][j] - nm);
        float ps = p0 + p1;
        ps += __shfl_xor(ps, 1); ps += __shfl_xor(ps, 2);
        ps += __shfl_xor(ps, 4); ps += __shfl_xor(ps, 8);
        rsum_s[j] = rsum_s[j] * corr + ps;
        acc_s0[j] *= corr; acc_s1[j] *= corr;
        Ps[row][l16]      = (__bf16)p0;
        Ps[row][16 + l16] = (__bf16)p1;
      }
      {
        float cm = fmaxf(sD[0][j], sD[1][j]);
        cm = fmaxf(cm, __shfl_xor(cm, 1)); cm = fmaxf(cm, __shfl_xor(cm, 2));
        cm = fmaxf(cm, __shfl_xor(cm, 4)); cm = fmaxf(cm, __shfl_xor(cm, 8));
        const float nm = fmaxf(rmax_d[j], cm);
        const float corr = __expf(rmax_d[j] - nm);
        rmax_d[j] = nm;
        const float p0 = __expf(sD[0][j] - nm);
        const float p1 = __expf(sD[1][j] - nm);
        float ps = p0 + p1;
        ps += __shfl_xor(ps, 1); ps += __shfl_xor(ps, 2);
        ps += __shfl_xor(ps, 4); ps += __shfl_xor(ps, 8);
        rsum_d[j] = rsum_d[j] * corr + ps;
        acc_d0[j] *= corr; acc_d1[j] *= corr;
        Pd[row][l16]      = (__bf16)p0;
        Pd[row][16 + l16] = (__bf16)p1;
      }
    }
    __syncthreads();

    // P A-fragments: two ds_load_b128 each
    const __bf16* psr = &Ps[l16][0];
    const __bf16* pdr = &Pd[l16][0];
    const v16bf pAs = ld_frag2(psr + half * 16, psr + half * 16 + 8);
    const v16bf pAd = ld_frag2(pdr + half * 16, pdr + half * 16 + 8);

    // V B-fragments: per lane 16 consecutive floats -> 4x global b128
    float vt0[16], vt1[16];
    const float* v0p = vp + (size_t)l16 * Np + mc + half * 16;
    const float* v1p = vp + (size_t)(16 + l16) * Np + mc + half * 16;
#pragma unroll
    for (int u = 0; u < 4; ++u) {
      *(float4*)&vt0[u * 4] = *(const float4*)(v0p + u * 4);
      *(float4*)&vt1[u * 4] = *(const float4*)(v1p + u * 4);
    }
    v16bf bv0, bv1;
#pragma unroll
    for (int i = 0; i < 16; ++i) {
      bv0[i] = (__bf16)vt0[i];
      bv1[i] = (__bf16)vt1[i];
    }

    acc_s0 = __builtin_amdgcn_wmma_f32_16x16x32_bf16(false, pAs, false, bv0, (short)0, acc_s0, false, false);
    acc_s1 = __builtin_amdgcn_wmma_f32_16x16x32_bf16(false, pAs, false, bv1, (short)0, acc_s1, false, false);
    acc_d0 = __builtin_amdgcn_wmma_f32_16x16x32_bf16(false, pAd, false, bv0, (short)0, acc_d0, false, false);
    acc_d1 = __builtin_amdgcn_wmma_f32_16x16x32_bf16(false, pAd, false, bv1, (short)0, acc_d1, false, false);
    __syncthreads();
  }

  // write ctx[b, h*HD + d, n]
#pragma unroll
  for (int j = 0; j < 8; ++j) {
    const int row = n0 + half * 8 + j;
    const float is = 1.f / rsum_s[j];
    const float id = 1.f / rsum_d[j];
    ctx_s[(size_t)b * CN + (size_t)(h * HDc + l16) * Np + row]      = acc_s0[j] * is;
    ctx_s[(size_t)b * CN + (size_t)(h * HDc + 16 + l16) * Np + row] = acc_s1[j] * is;
    ctx_d[(size_t)b * CN + (size_t)(h * HDc + l16) * Np + row]      = acc_d0[j] * id;
    ctx_d[(size_t)b * CN + (size_t)(h * HDc + 16 + l16) * Np + row] = acc_d1[j] * id;
  }
}

// ---------------------------------------------------------------------------
extern "C" void kernel_launch(void* const* d_in, const int* in_sizes, int n_in,
                              void* d_out, int out_size, void* d_ws, size_t ws_size,
                              hipStream_t stream) {
  (void)in_sizes; (void)n_in; (void)out_size; (void)ws_size;
  const float* xyzA   = (const float*)d_in[0];
  const float* featA  = (const float*)d_in[1];
  const float* xyzB   = (const float*)d_in[2];
  const float* featB  = (const float*)d_in[3];
  const float* ln_in_g = (const float*)d_in[4];
  const float* ln_in_b = (const float*)d_in[5];
  const float* Wq  = (const float*)d_in[6];   const float* bq  = (const float*)d_in[7];
  const float* Wk  = (const float*)d_in[8];   const float* bk_ = (const float*)d_in[9];
  const float* Wv  = (const float*)d_in[10];  const float* bv_ = (const float*)d_in[11];
  const float* sb_w1   = (const float*)d_in[12];
  const float* sb_bn_g = (const float*)d_in[13];
  const float* sb_bn_b = (const float*)d_in[14];
  const float* sb_w2   = (const float*)d_in[15];
  const float* sb_b2   = (const float*)d_in[16];
  const float* Wqd = (const float*)d_in[17];  const float* bqd = (const float*)d_in[18];
  const float* Wkd = (const float*)d_in[19];  const float* bkd = (const float*)d_in[20];
  const float* ln_sim_g = (const float*)d_in[21]; const float* ln_sim_b = (const float*)d_in[22];
  const float* ln_dis_g = (const float*)d_in[23]; const float* ln_dis_b = (const float*)d_in[24];
  const float* Wf1   = (const float*)d_in[25];
  const float* bnf_g = (const float*)d_in[26]; const float* bnf_b = (const float*)d_in[27];
  const float* Wf2 = (const float*)d_in[28];  const float* bf2 = (const float*)d_in[29];
  float* out = (float*)d_out;

  // workspace carve-up (floats)
  float* ws = (float*)d_ws;
  const size_t CN = (size_t)Cc * Np;
  size_t o = 0;
  float* fused = ws + o; o += (size_t)Bc * 3 * CN;   // [B,3C,N] concat buffer
  float* qbuf  = ws + o; o += (size_t)Bc * CN;
  float* kbuf  = ws + o; o += (size_t)Bc * CN;
  float* vbuf  = ws + o; o += (size_t)Bc * CN;
  float* qdb   = ws + o; o += (size_t)Bc * CN;
  float* kdb   = ws + o; o += (size_t)Bc * CN;
  float* ctxs  = ws + o; o += (size_t)Bc * CN;
  float* ctxd  = ws + o; o += (size_t)Bc * CN;
  float* f1b   = ws + o; o += (size_t)Bc * 2 * CN;
  float* dotd  = ws + o; o += (size_t)Bc * (size_t)Np * Np;
  float* nqv   = ws + o; o += (size_t)Bc * Np;
  float* nkv   = ws + o; o += (size_t)Bc * Np;

  const float bnrs = 0.99999500003749978f;
  const int cols = Bc * Np;

  // 1) featA LayerNorm -> slot 0 of fused buffer
  float* featAn = fused;
  ln_cn_kernel<<<dim3((cols + 7) / 8), 256, 0, stream>>>(
      featA, ln_in_g, ln_in_b, featAn, (long)CN, 3L * (long)CN, cols);

  // 2) projections (M=C, K=C, Nn=N per batch): shift-only epilogue
  auto proj = [&](const float* W, const float* X, float* Y, const float* bias,
                  long x_bs) {
    gemm_bf16_wmma<0, 0, 1, 0, 0><<<dim3(Np / 32, Cc / 64, Bc), 256, 0, stream>>>(
        W, X, Y, nullptr, bias, nullptr, Cc, Np, Cc,
        (long)Cc, 0L, (long)Np, x_bs, (long)CN, 0L, 1.0f);
  };
  proj(Wq,  featAn, qbuf, bq,  3L * (long)CN);
  proj(Wqd, featAn, qdb,  bqd, 3L * (long)CN);
  proj(Wk,  featB,  kbuf, bk_, (long)CN);
  proj(Wv,  featB,  vbuf, bv_, (long)CN);
  proj(Wkd, featB,  kdb,  bkd, (long)CN);

  // 3) feature-space dots: dot[b][m,n] = sum_c qd[b,c,m]*kd[b,c,n]  (A^T GEMM)
  gemm_bf16_wmma<1, 0, 0, 0, 0><<<dim3(Np / 32, Np / 64, Bc), 256, 0, stream>>>(
      qdb, kdb, dotd, nullptr, nullptr, nullptr, Np, Np, Cc,
      (long)Np, (long)CN, (long)Np, (long)CN, (long)Np * Np, 0L, 1.0f);

  // 4) squared norms for cdist expansion
  rownorm2_kernel<<<dim3(cols / 8), 256, 0, stream>>>(qdb, nqv, cols);
  rownorm2_kernel<<<dim3(cols / 8), 256, 0, stream>>>(kdb, nkv, cols);

  // 5) fused dual-path flash attention
  attn_dual_kernel<<<dim3(Np / 16, Hh, Bc), 32, 0, stream>>>(
      qbuf, kbuf, vbuf, dotd, nqv, nkv, xyzA, xyzB,
      sb_w1, sb_bn_g, sb_bn_b, sb_w2, sb_b2, ctxs, ctxd);

  // 6) LayerNorm contexts into fused slots 1 and 2
  ln_cn_kernel<<<dim3((cols + 7) / 8), 256, 0, stream>>>(
      ctxs, ln_sim_g, ln_sim_b, fused + CN, (long)CN, 3L * (long)CN, cols);
  ln_cn_kernel<<<dim3((cols + 7) / 8), 256, 0, stream>>>(
      ctxd, ln_dis_g, ln_dis_b, fused + 2 * CN, (long)CN, 3L * (long)CN, cols);

  // 7) fusion MLP: f = relu(BN(Wf1 @ fused)); out = Wf2 @ f + bf2 + featA
  gemm_bf16_wmma<0, 1, 1, 0, 1><<<dim3(Np / 32, (2 * Cc) / 64, Bc), 256, 0, stream>>>(
      Wf1, fused, f1b, bnf_g, bnf_b, nullptr, 2 * Cc, Np, 3 * Cc,
      3L * Cc, 0L, (long)Np, 3L * (long)CN, 2L * (long)CN, 0L, bnrs);
  gemm_bf16_wmma<0, 0, 1, 1, 0><<<dim3(Np / 32, Cc / 64, Bc), 256, 0, stream>>>(
      Wf2, f1b, out, nullptr, bf2, featA, Cc, Np, 2 * Cc,
      2L * Cc, 0L, (long)Np, 2L * (long)CN, (long)CN, (long)CN, 1.0f);
}
